// SAGELayer_85152021611247
// MI455X (gfx1250) — matrix operand order repaired and checked by reading, not compiled, not run on hardware
//
#include <hip/hip_runtime.h>
#include <hip/hip_bf16.h>

#define N_NODES 100000
#define N_EDGES 800000
#define D       128
#define KTOT    256
#define WAVES_PER_BLOCK 8

typedef __attribute__((ext_vector_type(16))) _Float16 v16h;
typedef __attribute__((ext_vector_type(8)))  float    v8f;

// ---------------------------------------------------------------------------
// Zero workspace (agg + deg) — must run every call (atomic accumulation).
// ---------------------------------------------------------------------------
__global__ void zero_kernel(float* __restrict__ p, size_t n) {
    size_t i = (size_t)blockIdx.x * blockDim.x + threadIdx.x;
    size_t stride = (size_t)gridDim.x * blockDim.x;
    for (; i < n; i += stride) p[i] = 0.0f;
}

// ---------------------------------------------------------------------------
// Shared helper: fill LDS with weight matrix (D x KTOT, row-major f32)
// pre-swizzled into per-lane WMMA B fragments.
// Fragment (kb, nt): 32 lanes x 16 f16 (32B per lane).
//   lane L -> column N = L&15, K = kb*32 + ((L<16)?0:8) + i + ((i>=8)?8:0), i=0..15
// 64 fragments * 32 lanes * 16 halves = 32768 halves = 64KB.
// ---------------------------------------------------------------------------
__device__ __forceinline__ void fill_w_lds(_Float16* wlds, const float* __restrict__ W) {
    for (int idx = threadIdx.x; idx < 64 * 32 * 16; idx += blockDim.x) {
        int f    = idx >> 9;          // fragment 0..63
        int rem  = idx & 511;
        int lane = rem >> 4;
        int i    = rem & 15;
        int kb   = f >> 3;
        int nt   = f & 7;
        int n    = lane & 15;
        int khalf = (lane < 16) ? 0 : 8;
        int k    = kb * 32 + khalf + i + ((i >= 8) ? 8 : 0);
        int o    = nt * 16 + n;
        wlds[idx] = (_Float16)W[o * KTOT + k];
    }
}

// Build A fragment (16x32 f16) for this lane from one 32-float segment of its source row.
// Elements 0..7  -> K = kbase + 0..7
// Elements 8..15 -> K = kbase + 16..23   (kbase = (lane<16)?0:8)
__device__ __forceinline__ v16h load_a_frag(const float* __restrict__ rp, int kbase, float scale) {
    const float4 x0 = *(const float4*)(rp + kbase + 0);
    const float4 x1 = *(const float4*)(rp + kbase + 4);
    const float4 x2 = *(const float4*)(rp + kbase + 16);
    const float4 x3 = *(const float4*)(rp + kbase + 20);
    v16h a;
    a[0]  = (_Float16)(x0.x * scale); a[1]  = (_Float16)(x0.y * scale);
    a[2]  = (_Float16)(x0.z * scale); a[3]  = (_Float16)(x0.w * scale);
    a[4]  = (_Float16)(x1.x * scale); a[5]  = (_Float16)(x1.y * scale);
    a[6]  = (_Float16)(x1.z * scale); a[7]  = (_Float16)(x1.w * scale);
    a[8]  = (_Float16)(x2.x * scale); a[9]  = (_Float16)(x2.y * scale);
    a[10] = (_Float16)(x2.z * scale); a[11] = (_Float16)(x2.w * scale);
    a[12] = (_Float16)(x3.x * scale); a[13] = (_Float16)(x3.y * scale);
    a[14] = (_Float16)(x3.z * scale); a[15] = (_Float16)(x3.w * scale);
    return a;
}

// ---------------------------------------------------------------------------
// Kernel A: per-edge message GEMM + fused segment-sum scatter (atomics) + deg.
// One wave = one 16-edge tile; 64 WMMA (8 N-tiles x 8 K-blocks) per tile.
// ---------------------------------------------------------------------------
__global__ void __launch_bounds__(256) edge_msg_kernel(
    const float* __restrict__ nfeats, const float* __restrict__ efeats,
    const int* __restrict__ src, const int* __restrict__ dst,
    const float* __restrict__ Wm, const float* __restrict__ bm,
    float* __restrict__ agg, float* __restrict__ deg)
{
    __shared__ __align__(32) _Float16 wlds[64 * 32 * 16];   // 64KB
    fill_w_lds(wlds, Wm);
    __syncthreads();

    const int lane   = threadIdx.x & 31;
    const int waveId = blockIdx.x * WAVES_PER_BLOCK + (threadIdx.x >> 5);
    const int nWaves = gridDim.x * WAVES_PER_BLOCK;
    const int nlo    = lane & 15;
    const int kbase  = (lane < 16) ? 0 : 8;
    const int hi     = (lane < 16) ? 0 : 8;

    for (int tile = waveId; tile < N_EDGES / 16; tile += nWaves) {
        const int base = tile * 16;
        const int m    = base + nlo;                     // edge id for this lane's A row
        const int s    = src[m];
        const float* hrow = nfeats + (size_t)s * D;
        const float* erow = efeats + (size_t)m * D;

        v8f acc[8];
        #pragma unroll
        for (int nt = 0; nt < 8; ++nt) {
            const float bv = bm[nt * 16 + nlo];          // bias for output o = nt*16 + N
            #pragma unroll
            for (int j = 0; j < 8; ++j) acc[nt][j] = bv;
        }

        #pragma unroll
        for (int kb = 0; kb < 8; ++kb) {
            const float* rp = (kb < 4) ? (hrow + kb * 32) : (erow + (kb - 4) * 32);
            const v16h a = load_a_frag(rp, kbase, 1.0f);
            const _Float16* wp = &wlds[((size_t)(kb * 8) * 32 + lane) * 16];
            #pragma unroll
            for (int nt = 0; nt < 8; ++nt) {
                const v16h bfrag = *(const v16h*)(wp + (size_t)nt * 32 * 16);
                acc[nt] = __builtin_amdgcn_wmma_f32_16x16x32_f16(
                    false, a, false, bfrag, (short)0, acc[nt], false, false);
            }
        }

        // Fused segment_sum: scatter-add into agg[dst[e]].
        #pragma unroll
        for (int j = 0; j < 8; ++j) {
            const int e = base + hi + j;                 // row M = hi + j
            const int dnode = dst[e];
            float* arow = agg + (size_t)dnode * D;
            #pragma unroll
            for (int nt = 0; nt < 8; ++nt) {
                unsafeAtomicAdd(arow + nt * 16 + nlo, acc[nt][j]);
            }
        }
        if (lane < 16) {
            unsafeAtomicAdd(deg + dst[m], 1.0f);
        }
    }
}

// ---------------------------------------------------------------------------
// Kernel B: node apply GEMM: new_h = relu(W_apply * concat(h, agg/deg) + b)
// ---------------------------------------------------------------------------
__global__ void __launch_bounds__(256) node_apply_kernel(
    const float* __restrict__ nfeats, const float* __restrict__ agg,
    const float* __restrict__ deg,
    const float* __restrict__ Wa, const float* __restrict__ ba,
    float* __restrict__ new_h)
{
    __shared__ __align__(32) _Float16 wlds[64 * 32 * 16];   // 64KB
    fill_w_lds(wlds, Wa);
    __syncthreads();

    const int lane   = threadIdx.x & 31;
    const int waveId = blockIdx.x * WAVES_PER_BLOCK + (threadIdx.x >> 5);
    const int nWaves = gridDim.x * WAVES_PER_BLOCK;
    const int nlo    = lane & 15;
    const int kbase  = (lane < 16) ? 0 : 8;
    const int hi     = (lane < 16) ? 0 : 8;

    for (int tile = waveId; tile < N_NODES / 16; tile += nWaves) {
        const int base = tile * 16;
        const int m    = base + nlo;                     // node id for this lane's A row
        const float* hrow = nfeats + (size_t)m * D;
        const float* arow = agg    + (size_t)m * D;
        const float inv = 1.0f / fmaxf(deg[m], 1.0f);

        v8f acc[8];
        #pragma unroll
        for (int nt = 0; nt < 8; ++nt) {
            const float bv = ba[nt * 16 + nlo];
            #pragma unroll
            for (int j = 0; j < 8; ++j) acc[nt][j] = bv;
        }

        #pragma unroll
        for (int kb = 0; kb < 8; ++kb) {
            const float* rp   = (kb < 4) ? (hrow + kb * 32) : (arow + (kb - 4) * 32);
            const float scale = (kb < 4) ? 1.0f : inv;   // h_neigh = agg / max(deg,1)
            const v16h a = load_a_frag(rp, kbase, scale);
            const _Float16* wp = &wlds[((size_t)(kb * 8) * 32 + lane) * 16];
            #pragma unroll
            for (int nt = 0; nt < 8; ++nt) {
                const v16h bfrag = *(const v16h*)(wp + (size_t)nt * 32 * 16);
                acc[nt] = __builtin_amdgcn_wmma_f32_16x16x32_f16(
                    false, a, false, bfrag, (short)0, acc[nt], false, false);
            }
        }

        // ReLU + store
        #pragma unroll
        for (int nt = 0; nt < 8; ++nt) {
            const int o = nt * 16 + nlo;
            #pragma unroll
            for (int j = 0; j < 8; ++j) {
                const int node = base + hi + j;
                new_h[(size_t)node * D + o] = fmaxf(acc[nt][j], 0.0f);
            }
        }
    }
}

// ---------------------------------------------------------------------------
// Kernel C: new_e = 0.5 * (new_h[src] + new_h[dst]); float4 vectorized.
// 32 threads per edge, 4 floats each.
// ---------------------------------------------------------------------------
__global__ void __launch_bounds__(256) edge_out_kernel(
    const float* __restrict__ new_h, const int* __restrict__ src,
    const int* __restrict__ dst, float* __restrict__ new_e)
{
    const size_t tid = (size_t)blockIdx.x * blockDim.x + threadIdx.x;
    const int e = (int)(tid >> 5);
    const int c = (int)(tid & 31) << 2;
    if (e < N_EDGES) {
        const int s = src[e], d = dst[e];
        const float4 a = *(const float4*)(new_h + (size_t)s * D + c);
        const float4 b = *(const float4*)(new_h + (size_t)d * D + c);
        float4 r;
        r.x = (a.x + b.x) * 0.5f;
        r.y = (a.y + b.y) * 0.5f;
        r.z = (a.z + b.z) * 0.5f;
        r.w = (a.w + b.w) * 0.5f;
        *(float4*)(new_e + (size_t)e * D + c) = r;
    }
}

// ---------------------------------------------------------------------------
extern "C" void kernel_launch(void* const* d_in, const int* in_sizes, int n_in,
                              void* d_out, int out_size, void* d_ws, size_t ws_size,
                              hipStream_t stream) {
    const float* nfeats = (const float*)d_in[0];
    const float* efeats = (const float*)d_in[1];
    const int*   src    = (const int*)d_in[2];
    const int*   dst    = (const int*)d_in[3];
    const float* Wm     = (const float*)d_in[4];
    const float* bm     = (const float*)d_in[5];
    const float* Wa     = (const float*)d_in[6];
    const float* ba     = (const float*)d_in[7];

    float* new_h = (float*)d_out;                        // N_NODES * D
    float* new_e = new_h + (size_t)N_NODES * D;          // N_EDGES * D

    float* agg = (float*)d_ws;                           // N_NODES * D
    float* deg = agg + (size_t)N_NODES * D;              // N_NODES

    const size_t zero_n = (size_t)N_NODES * D + N_NODES;
    zero_kernel<<<2048, 256, 0, stream>>>(agg, zero_n);

    edge_msg_kernel<<<1250, 256, 0, stream>>>(nfeats, efeats, src, dst, Wm, bm, agg, deg);

    node_apply_kernel<<<625, 256, 0, stream>>>(nfeats, agg, deg, Wa, ba, new_h);

    const long long c_threads = (long long)N_EDGES * 32;
    edge_out_kernel<<<(int)((c_threads + 255) / 256), 256, 0, stream>>>(new_h, src, dst, new_e);
}
